// SpatialGraphModule_66245575573972
// MI455X (gfx1250) — matrix operand order repaired and checked by reading, not compiled, not run on hardware
//
#include <hip/hip_runtime.h>
#include <hip/hip_bf16.h>
#include <math.h>

// ---------------------------------------------------------------------------
// SpatialGraphModule, fused for MI455X (gfx1250, wave32, WMMA + TDM).
//
// Exact simplification: LayerNorm over a singleton dim makes every adjacency
// logit equal -> softmax uniform -> adj == 1/(6+1e-8).  Hence P and P^2 of
// the Chebyshev recurrence are compile-time constants of the form
// po*ones + (pd-po)*I, and nmean = u * colsum(g).  All heavy work reduces to
// three dense GEMMs over 49152 rows, executed with v_wmma_f32_16x16x32_f16.
// Weight tiles are pre-converted to f16 (N,K) once and DMA'd into LDS with
// the Tensor Data Mover (tensor_load_to_lds + s_wait_tensorcnt).
// ---------------------------------------------------------------------------

typedef __attribute__((ext_vector_type(16))) _Float16 v16h;
typedef __attribute__((ext_vector_type(8)))  float    v8f;

#define B_   64
#define C_   6
#define T_   128
#define H_   128
#define BT_  (B_ * T_)        // 8192
#define ROWS (BT_ * C_)       // 49152

// adjacency / Chebyshev constants (exact, see analysis)
constexpr double Ud   = 1.0 / (6.0 + 1e-8);
constexpr double RSd  = 1.0 + 6.0 * Ud;            // rowsum of A_hat
constexpr double PDd  = (1.0 + Ud) / RSd;          // P diagonal
constexpr double POd  = Ud / RSd;                  // P off-diagonal
constexpr float  UADJ   = (float)Ud;
constexpr float  P1_OFF = (float)POd;
constexpr float  P1_DIF = (float)(PDd - POd);
constexpr float  P2_OFF = (float)(2.0 * PDd * POd + 4.0 * POd * POd);
constexpr float  P2_DIF = (float)((PDd * PDd + 5.0 * POd * POd) -
                                  (2.0 * PDd * POd + 4.0 * POd * POd));

union Frag { v16h v; unsigned u[8]; };

#if defined(__has_builtin)
#if __has_builtin(__builtin_amdgcn_tensor_load_to_lds) && \
    __has_builtin(__builtin_amdgcn_s_wait_tensorcnt)
#define HAVE_TDM 1
#endif
#endif

__device__ __forceinline__ float gelu_exact(float x) {
    return 0.5f * x * (1.0f + erff(x * 0.70710678118654752f));
}

#ifdef HAVE_TDM
// One-shot TDM copy: 32KB contiguous (pre-transposed f16 weight tile) from
// global into LDS.  1-D descriptor: count=1, data_size=4B, tile_dim0=8192,
// type=2 ("image").  Issued by one wave; EXEC is ignored by TENSOR ops.
__device__ __forceinline__ void tdm_load_b(const _Float16* src, void* ldsDst) {
    typedef __attribute__((ext_vector_type(4))) unsigned u32x4;
    typedef __attribute__((ext_vector_type(8))) int      i32x8;
    typedef __attribute__((ext_vector_type(4))) int      i32x4;
    const unsigned long long ga = (unsigned long long)src;
    // flat shared pointer: addr[31:0] is the LDS byte offset (aperture map)
    const unsigned lds = (unsigned)(unsigned long long)ldsDst;
    u32x4 g0 = {};
    g0[0] = 1u;                                        // count=1 (valid D#)
    g0[1] = lds;                                       // lds_addr
    g0[2] = (unsigned)(ga & 0xffffffffu);              // global_addr[31:0]
    g0[3] = (unsigned)((ga >> 32) & 0x1ffffffu)        // global_addr[56:32]
            | (2u << 30);                              // type=2
    i32x8 g1 = {};
    g1[0] = (int)(2u << 16);        // workgroup_mask=0, data_size=2 (4B)
    g1[1] = (int)(8192u << 16);     // tensor_dim0[15:0]=8192 (bits 63:48)
    g1[2] = (int)(1u << 16);        // tensor_dim0[31:16]=0; tensor_dim1=1
    g1[3] = (int)(8192u << 16);     // tile_dim0=8192 (bits 127:112)
    g1[4] = 0;                      // tile_dim1=0, tile_dim2=0 (1-D tile)
    g1[5] = 8192;                   // tensor_dim0_stride
    g1[6] = 0; g1[7] = 0;
    i32x4 gz = {};
#if defined(__clang_major__) && __clang_major__ >= 23
    i32x8 gz8 = {};
    __builtin_amdgcn_tensor_load_to_lds(g0, g1, gz, gz, gz8, 0);
#else
    __builtin_amdgcn_tensor_load_to_lds(g0, g1, gz, gz, 0);
#endif
}
#endif

// Stage a pre-converted (N,K) f16 weight tile (32KB) into LDS.
__device__ __forceinline__ void stage_b(const _Float16* wsrc, _Float16* ldsBt,
                                        int tid) {
#ifdef HAVE_TDM
    if (tid == 0) {
        tdm_load_b(wsrc, (void*)ldsBt);
        __builtin_amdgcn_s_wait_tensorcnt(0);
    }
#else
    const uint4* s4 = (const uint4*)wsrc;
    uint4*       d4 = (uint4*)ldsBt;
#pragma unroll
    for (int j = 0; j < 8; ++j) d4[j * 256 + tid] = s4[j * 256 + tid];
#endif
}

// WMMA core: one 128-K chunk.  ldsA is 128x128 f16 row-major (M,K),
// ldsBt is 128x128 f16 stored (N,K) so fragment dwords are contiguous.
// Each wave (8 per block) owns rows [m0, m0+16) x all 128 cols -> acc[8].
// All 8 B fragments are preloaded per k-step so the 8 WMMAs issue while
// dscnt drains (software pipelining instead of s_wait_dscnt 0 per WMMA).
__device__ __forceinline__ void wmma_chunk(const _Float16* ldsA,
                                           const _Float16* ldsBt,
                                           v8f* acc, int lane, int m0) {
    const int li = lane & 15;
    const int hi = lane >> 4;
#pragma unroll
    for (int kk = 0; kk < 4; ++kk) {
        // A fragment: 16x32 f16 per ISA layout (lanes 0-15: K 0-7/16-23,
        // lanes 16-31: K 8-15/24-31)
        const _Float16* ap = ldsA + (m0 + li) * 128 + kk * 32 + hi * 8;
        Frag a;
#pragma unroll
        for (int v = 0; v < 8; ++v) {
            const int k0 = (v < 4) ? (2 * v) : (16 + 2 * (v - 4));
            a.u[v] = *(const unsigned*)(ap + k0);
        }
        // Preload all 8 B fragments (same K order as A: lanes 0-15: K 0-7/16-23, lanes 16-31: K 8-15/24-31)
        Frag b[8];
#pragma unroll
        for (int n = 0; n < 8; ++n) {
            const _Float16* bp = ldsBt + (n * 16 + li) * 128 + kk * 32 + hi * 8;
#pragma unroll
            for (int v = 0; v < 8; ++v) b[n].u[v] = *(const unsigned*)(bp + ((v < 4) ? (2 * v) : (16 + 2 * (v - 4))));
        }
#pragma unroll
        for (int n = 0; n < 8; ++n)
            acc[n] = __builtin_amdgcn_wmma_f32_16x16x32_f16(
                false, a.v, false, b[n].v, (short)0, acc[n], false, false);
    }
}

__device__ __forceinline__ void store_acc(float* ldsOut, const v8f* acc,
                                          int lane, int m0) {
    const int li = lane & 15;
    const int ro = (lane >> 4) * 8;   // C/D layout: lanes 16-31 hold rows +8
#pragma unroll
    for (int n = 0; n < 8; ++n)
#pragma unroll
        for (int j = 0; j < 8; ++j)
            ldsOut[(m0 + ro + j) * 128 + n * 16 + li] = acc[n][j];
}

// ---------------------------------------------------------------------------
// Kernel 0: pre-convert all weight chunks to f16, (N,K)-transposed, into ws.
// chunk 0 = W1, chunks 1..3 = cheb_w[k], chunks 4..5 = sW1 halves.
// ---------------------------------------------------------------------------
__global__ __launch_bounds__(256) void wprep_kernel(
    const float* __restrict__ W1, const float* __restrict__ Wc,
    const float* __restrict__ Ws1, _Float16* __restrict__ Wt) {
    const int e = blockIdx.x * 256 + threadIdx.x;      // < 6*16384
    const int chunk = e >> 14;
    const int idx = e & 16383;
    const int k = idx >> 7, n = idx & 127;
    const float* src = (chunk == 0) ? W1
                     : (chunk <= 3) ? Wc + (chunk - 1) * 16384
                                    : Ws1 + (chunk - 4) * 16384;
    Wt[chunk * 16384 + n * 128 + k] = (_Float16)src[k * 128 + n];
}

// ---------------------------------------------------------------------------
// Kernel 1: fe_proj.  Xn = gelu(LN(X @ W1 + b1)).  A rows gather the
// (b,C,T,F) -> (b,T,C,F) transpose; rows are contiguous in F.
// ---------------------------------------------------------------------------
__global__ __launch_bounds__(256) void fe_kernel(
    const float* __restrict__ X, const _Float16* __restrict__ Wt,
    const float* __restrict__ bias, const float* __restrict__ lng,
    const float* __restrict__ lnb, float* __restrict__ Xn) {
    __shared__ __align__(16) char raw[65536];
    _Float16* ldsA  = (_Float16*)raw;
    _Float16* ldsBt = (_Float16*)(raw + 32768);
    float*    ldsOut = (float*)raw;          // overlays A/B after compute

    const int tid  = threadIdx.x;
    const int row0 = blockIdx.x * 128;

    stage_b(Wt, ldsBt, tid);                 // TDM DMA of the weight tile
#pragma unroll 4
    for (int j = 0; j < 64; ++j) {           // stage A (f32 -> f16)
        const int e = j * 256 + tid;
        const int r = e >> 7, k = e & 127;
        const int gr = row0 + r;
        const int bt = gr / C_, c = gr - bt * C_;
        const int b  = bt >> 7, t = bt & 127;
        ldsA[r * 128 + k] =
            (_Float16)X[(((b * C_ + c) << 7) + t) * 128 + k];
    }
    __syncthreads();

    v8f acc[8] = {};
    const int lane = tid & 31, m0 = (tid >> 5) * 16;
    wmma_chunk(ldsA, ldsBt, acc, lane, m0);
    __syncthreads();
    store_acc(ldsOut, acc, lane, m0);
    __syncthreads();

    if (tid < 128) {                         // row epilogue: +b1, LN, GELU
        const int gr = row0 + tid;
        const float* orow = ldsOut + tid * 128;
        float s = 0.f, ss = 0.f;
        for (int c = 0; c < 128; ++c) {
            const float x = orow[c] + bias[c];
            s += x; ss += x * x;
        }
        const float m  = s * (1.f / 128.f);
        const float vr = ss * (1.f / 128.f) - m * m;
        const float rstd = rsqrtf(vr + 1e-5f);
        for (int c = 0; c < 128; ++c) {
            const float x = orow[c] + bias[c];
            Xn[gr * 128 + c] = gelu_exact((x - m) * rstd * lng[c] + lnb[c]);
        }
    }
}

// ---------------------------------------------------------------------------
// Kernel 2: adjacency output (uniform by the singleton-LN collapse)
// ---------------------------------------------------------------------------
__global__ __launch_bounds__(256) void adjfill_kernel(float* __restrict__ adj, int n) {
    const int i = blockIdx.x * 256 + threadIdx.x;
    if (i < n) adj[i] = UADJ;
}

// ---------------------------------------------------------------------------
// Kernel 3: per-bt column sums over the 6 nodes (feeds P*X and nmean)
// ---------------------------------------------------------------------------
__global__ __launch_bounds__(256) void colsum6_kernel(
    const float* __restrict__ src, float* __restrict__ dst, int n) {
    const int i = blockIdx.x * 256 + threadIdx.x;
    if (i >= n) return;
    const int bt = i >> 7, k = i & 127;
    const float* p = src + (bt * C_) * 128 + k;
    float s = 0.f;
#pragma unroll
    for (int c = 0; c < C_; ++c) s += p[c * 128];
    dst[i] = s;
}

// ---------------------------------------------------------------------------
// Kernel 4: ChebConv.  out = X@w0 + (PX)@w1 + (P^2 X)@w2, +b, LN -> g.
// K=384 as 3 chunks; hop-s A tiles are built on the fly from X and colsum.
// ---------------------------------------------------------------------------
__global__ __launch_bounds__(256) void cheb_kernel(
    const float* __restrict__ Xf, const float* __restrict__ Sx,
    const _Float16* __restrict__ Wt, const float* __restrict__ bias,
    const float* __restrict__ lng, const float* __restrict__ lnb,
    float* __restrict__ g) {
    __shared__ __align__(16) char raw[65536];
    _Float16* ldsA  = (_Float16*)raw;
    _Float16* ldsBt = (_Float16*)(raw + 32768);
    float*    ldsOut = (float*)raw;

    const int tid  = threadIdx.x;
    const int row0 = blockIdx.x * 128;
    const int lane = tid & 31, m0 = (tid >> 5) * 16;
    v8f acc[8] = {};

    for (int s = 0; s < 3; ++s) {
        const float off = (s == 0) ? 0.f : (s == 1 ? P1_OFF : P2_OFF);
        const float dif = (s == 0) ? 1.f : (s == 1 ? P1_DIF : P2_DIF);
        stage_b(Wt + s * 16384, ldsBt, tid);
#pragma unroll 2
        for (int j = 0; j < 64; ++j) {       // stage A_s
            const int e = j * 256 + tid;
            const int r = e >> 7, k = e & 127;
            const int gr = row0 + r;
            const int bt = gr / C_;
            const float base = Xf[gr * 128 + k];
            const float a = (s == 0) ? base : off * Sx[bt * 128 + k] + dif * base;
            ldsA[r * 128 + k] = (_Float16)a;
        }
        __syncthreads();
        wmma_chunk(ldsA, ldsBt, acc, lane, m0);
        __syncthreads();                     // before LDS reuse next chunk
    }
    store_acc(ldsOut, acc, lane, m0);
    __syncthreads();

    if (tid < 128) {                         // +cheb_b, LN -> g (no GELU)
        const int gr = row0 + tid;
        const float* orow = ldsOut + tid * 128;
        float s = 0.f, ss = 0.f;
        for (int c = 0; c < 128; ++c) {
            const float x = orow[c] + bias[c];
            s += x; ss += x * x;
        }
        const float m  = s * (1.f / 128.f);
        const float vr = ss * (1.f / 128.f) - m * m;
        const float rstd = rsqrtf(vr + 1e-5f);
        for (int c = 0; c < 128; ++c) {
            const float x = orow[c] + bias[c];
            g[gr * 128 + c] = (x - m) * rstd * lng[c] + lnb[c];
        }
    }
}

// ---------------------------------------------------------------------------
// Kernel 5: satt.  h = gelu(LN([g | u*colsum(g)] @ sW1 + sb1)),
// att = sigmoid(h @ sW2 + sb2) computed in the epilogue (h stays in LDS).
// ---------------------------------------------------------------------------
__global__ __launch_bounds__(256) void satt_kernel(
    const float* __restrict__ g, const float* __restrict__ Sg,
    const _Float16* __restrict__ Wt, const float* __restrict__ b1s,
    const float* __restrict__ lng, const float* __restrict__ lnb,
    const float* __restrict__ W2s, const float* __restrict__ b2s,
    float* __restrict__ att) {
    __shared__ __align__(16) char raw[65536];
    _Float16* ldsA  = (_Float16*)raw;
    _Float16* ldsBt = (_Float16*)(raw + 32768);
    float*    ldsOut = (float*)raw;

    const int tid  = threadIdx.x;
    const int row0 = blockIdx.x * 128;
    const int lane = tid & 31, m0 = (tid >> 5) * 16;
    v8f acc[8] = {};

    for (int s = 0; s < 2; ++s) {
        stage_b(Wt + s * 16384, ldsBt, tid);
#pragma unroll 2
        for (int j = 0; j < 64; ++j) {       // stage A: g, then nmean
            const int e = j * 256 + tid;
            const int r = e >> 7, k = e & 127;
            const int gr = row0 + r;
            const int bt = gr / C_;
            const float a = (s == 0) ? g[gr * 128 + k]
                                     : UADJ * Sg[bt * 128 + k];
            ldsA[r * 128 + k] = (_Float16)a;
        }
        __syncthreads();
        wmma_chunk(ldsA, ldsBt, acc, lane, m0);
        __syncthreads();
    }
    store_acc(ldsOut, acc, lane, m0);
    __syncthreads();

    if (tid < 128) {                         // +sb1, LN, GELU, dot sW2, sigmoid
        const int gr = row0 + tid;
        const float* orow = ldsOut + tid * 128;
        float s = 0.f, ss = 0.f;
        for (int c = 0; c < 128; ++c) {
            const float x = orow[c] + b1s[c];
            s += x; ss += x * x;
        }
        const float m  = s * (1.f / 128.f);
        const float vr = ss * (1.f / 128.f) - m * m;
        const float rstd = rsqrtf(vr + 1e-5f);
        float dot = 0.f;
        for (int c = 0; c < 128; ++c) {
            const float x = orow[c] + b1s[c];
            const float h = gelu_exact((x - m) * rstd * lng[c] + lnb[c]);
            dot += h * W2s[c];
        }
        att[gr] = 1.f / (1.f + expf(-(dot + b2s[0])));
    }
}

// ---------------------------------------------------------------------------
// Kernel 6: epoch = mean over nodes of g * att
// ---------------------------------------------------------------------------
__global__ __launch_bounds__(256) void epoch_kernel(
    const float* __restrict__ g, const float* __restrict__ att,
    float* __restrict__ epoch, int n) {
    const int i = blockIdx.x * 256 + threadIdx.x;
    if (i >= n) return;
    const int bt = i >> 7, f = i & 127;
    float s = 0.f;
#pragma unroll
    for (int c = 0; c < C_; ++c)
        s += g[(bt * C_ + c) * 128 + f] * att[bt * C_ + c];
    epoch[i] = s * (1.f / (float)C_);
}

// ---------------------------------------------------------------------------
extern "C" void kernel_launch(void* const* d_in, const int* in_sizes, int n_in,
                              void* d_out, int out_size, void* d_ws, size_t ws_size,
                              hipStream_t stream) {
    (void)in_sizes; (void)n_in; (void)out_size; (void)ws_size;
    const float* X_de  = (const float*)d_in[0];
    const float* W1    = (const float*)d_in[1];
    const float* b1    = (const float*)d_in[2];
    const float* ln1g  = (const float*)d_in[3];
    const float* ln1b  = (const float*)d_in[4];
    // d_in[5..8] (aw, ab, aln_g, aln_b) provably unused: LN over a singleton
    // dim collapses to its bias, so adjacency logits are constant.
    const float* chebw = (const float*)d_in[9];
    const float* chebb = (const float*)d_in[10];
    const float* clng  = (const float*)d_in[11];
    const float* clnb  = (const float*)d_in[12];
    const float* sW1   = (const float*)d_in[13];
    const float* sb1   = (const float*)d_in[14];
    const float* slng  = (const float*)d_in[15];
    const float* slnb  = (const float*)d_in[16];
    const float* sW2   = (const float*)d_in[17];
    const float* sb2   = (const float*)d_in[18];

    float* out   = (float*)d_out;
    float* epoch = out;                              // (B,T,H)   1048576
    float* adj   = out + BT_ * H_;                   // (B,T,C,C)  294912
    float* Xn    = adj + BT_ * C_ * C_;              // (B,T,C,H) 6291456
    float* gout  = Xn + ROWS * H_;                   // (B,T,C,H) 6291456

    float* ws  = (float*)d_ws;
    float* Sx  = ws;                                 // (BT,H) colsum of Xn
    float* Sg  = ws + BT_ * H_;                      // (BT,H) colsum of g
    float* att = ws + 2 * BT_ * H_;                  // (ROWS)
    _Float16* Wt = (_Float16*)(ws + 2 * BT_ * H_ + ROWS); // 6x16384 f16 tiles

    const int gemmBlocks = ROWS / 128;               // 384
    const int btH = BT_ * H_;                        // 1048576

    wprep_kernel<<<(6 * 16384) / 256, 256, 0, stream>>>(W1, chebw, sW1, Wt);
    fe_kernel<<<gemmBlocks, 256, 0, stream>>>(X_de, Wt, b1, ln1g, ln1b, Xn);
    adjfill_kernel<<<(BT_ * C_ * C_ + 255) / 256, 256, 0, stream>>>(adj, BT_ * C_ * C_);
    colsum6_kernel<<<btH / 256, 256, 0, stream>>>(Xn, Sx, btH);
    cheb_kernel<<<gemmBlocks, 256, 0, stream>>>(Xn, Sx, Wt + 16384, chebb, clng, clnb, gout);
    colsum6_kernel<<<btH / 256, 256, 0, stream>>>(gout, Sg, btH);
    satt_kernel<<<gemmBlocks, 256, 0, stream>>>(gout, Sg, Wt + 4 * 16384, sb1, slng, slnb, sW2, sb2, att);
    epoch_kernel<<<btH / 256, 256, 0, stream>>>(gout, att, epoch, btH);
}
